// DecoderLayer_5514738008248
// MI455X (gfx1250) — matrix operand order repaired and checked by reading, not compiled, not run on hardware
//
#include <hip/hip_runtime.h>
#include <stdint.h>

// ---------------------------------------------------------------------------
// CDNA5 (gfx1250) transformer decoder layer, bf16 WMMA everywhere.
// B=2, S=T=2048, E=1024, H=16, HD=64, F=4096.
// ---------------------------------------------------------------------------

typedef unsigned short u16;
typedef __attribute__((ext_vector_type(16))) __bf16 v16bf;
typedef __attribute__((ext_vector_type(8)))  float  v8f;

#define B_NUM 2
#define S_LEN 2048
#define T_LEN 2048
#define E_DIM 1024
#define H_NUM 16
#define HD_DIM 64
#define F_DIM 4096
#define M_ROWS (B_NUM * S_LEN)   // 4096 flattened rows

__device__ __forceinline__ u16 f32_to_bf16_rne(float f) {
  union { float f; unsigned int u; } v; v.f = f;
  unsigned int u = v.u;
  return (u16)((u + 0x7FFFu + ((u >> 16) & 1u)) >> 16);
}

// ------------------------------ f32 -> bf16 --------------------------------
__global__ void cvt_f32_to_bf16(const float* __restrict__ in,
                                u16* __restrict__ out, int n) {
  int i = blockIdx.x * 256 + threadIdx.x;
  if (i < n) out[i] = f32_to_bf16_rne(in[i]);
}

// ------------------------------ WMMA GEMM ----------------------------------
// C[M,N] = A[M,K] * W[N,K]^T + bias (torch Linear layout: both K-contiguous).
// Block: 256 thr = 8 waves (2x4); block tile 128x128; wave tile 64x32.
// Staging: gfx1250 GLOBAL_LOAD_ASYNC_TO_LDS_B128 (no VGPR residency, ASYNCcnt
// tracked) into double-buffered, fragment-pre-swizzled LDS:
//   A-frag (16x32 bf16): lane l (l<16) row=l, K={0..7,16..23}; lane>=16 K={8..15,24..31}
//   B-frag (32x16 bf16): lane l col=l&15, K={0..7,16..23} (l<16) / {8..15,24..31} (l>=16)
// so each lane's fragment is 16 contiguous bf16 (32B) in LDS, and each lane's
// 16B global chunk maps to exactly one 16B LDS slot (async per-lane transfer).
__global__ __launch_bounds__(256)
void gemm_bf16_wmma(const u16* __restrict__ A, const u16* __restrict__ W,
                    const float* __restrict__ bias,
                    float* __restrict__ outF,      // optional [M,N] f32
                    u16*   __restrict__ outB,      // optional bf16 (plain or head-split)
                    u16*   __restrict__ outBT,     // optional bf16 [B,H,HD,seq] (V^T)
                    int M, int N, int K, int relu, int head_mode)
{
  __shared__ __align__(32) u16 As[2 * 4096];  // 2 bufs x (8 sub-blk x 32 ln x 16)
  __shared__ __align__(32) u16 Bs[2 * 4096];

  const int tid  = threadIdx.x;
  const int lane = tid & 31;
  const int wid  = tid >> 5;
  const int wrow = wid >> 2;          // 0..1
  const int wcol = wid & 3;           // 0..3
  const int m0 = blockIdx.y * 128;
  const int n0 = blockIdx.x * 128;

  // Per-thread staging tasks: 512 tasks = 128 rows x 4 chunks of 8 bf16.
  const u16* aptr[2];
  const u16* wptr[2];
  int ldsA[2], ldsB[2];
#pragma unroll
  for (int it = 0; it < 2; ++it) {
    const int task = tid + it * 256;
    const int r = task >> 2;
    const int c = task & 3;
    aptr[it] = A + (size_t)(m0 + r) * K + 8 * c;
    wptr[it] = W + (size_t)(n0 + r) * K + 8 * c;
    ldsA[it] = (r >> 4) * 512 + ((r & 15) | ((c & 1) << 4)) * 16 + ((c >> 1) << 3);
    ldsB[it] = (r >> 4) * 512 + ((r & 15) | ((c & 1) << 4)) * 16 + ((c >> 1) << 3);
  }

  // Async global->LDS stage of one 128x32 A tile + 128x32 W tile (16B/lane).
  auto stage_async = [&](int k, int bofs) {
#pragma unroll
    for (int it = 0; it < 2; ++it) {
      const unsigned la = (unsigned)(uintptr_t)(&As[bofs + ldsA[it]]);
      const unsigned lb = (unsigned)(uintptr_t)(&Bs[bofs + ldsB[it]]);
      asm volatile("global_load_async_to_lds_b128 %0, %1, off"
                   :: "v"(la), "v"((const void*)(aptr[it] + k)) : "memory");
      asm volatile("global_load_async_to_lds_b128 %0, %1, off"
                   :: "v"(lb), "v"((const void*)(wptr[it] + k)) : "memory");
    }
  };

  v8f acc[4][2];
#pragma unroll
  for (int i = 0; i < 4; ++i)
#pragma unroll
    for (int j = 0; j < 2; ++j)
#pragma unroll
      for (int e = 0; e < 8; ++e) acc[i][j][e] = 0.0f;

  // Pipeline prologue: k-step 0 in flight.
  stage_async(0, 0);

  int buf = 0;
  for (int k0 = 0; k0 < K; k0 += 32) {
    const int bofs = buf << 12;  // 4096 u16 per buffer
    // Wait for our own async transfers, then make all waves' LDS visible.
    asm volatile("s_wait_asynccnt 0x0" ::: "memory");
    __syncthreads();
    // Issue next k-step (other buffer) -- overlaps the whole compute phase.
    if (k0 + 32 < K) stage_async(k0 + 32, bofs ^ 4096);
    // -------- fragments + 8x WMMA ------------------------------------------
    v16bf af[4], bfr[2];
#pragma unroll
    for (int i = 0; i < 4; ++i)
      af[i] = *(const v16bf*)(&As[bofs + (wrow * 4 + i) * 512 + lane * 16]);
#pragma unroll
    for (int j = 0; j < 2; ++j)
      bfr[j] = *(const v16bf*)(&Bs[bofs + (wcol * 2 + j) * 512 + lane * 16]);
#pragma unroll
    for (int i = 0; i < 4; ++i)
#pragma unroll
      for (int j = 0; j < 2; ++j)
        acc[i][j] = __builtin_amdgcn_wmma_f32_16x16x32_bf16(
            false, af[i], false, bfr[j], (short)0, acc[i][j], false, false);
    buf ^= 1;
  }

  // -------- epilogue: bias (+relu), f32 / bf16 / head-split / V^T stores ---
  const int half = lane >> 4;
  const int nlo  = lane & 15;
#pragma unroll
  for (int j = 0; j < 2; ++j) {
    const int n = n0 + wcol * 32 + j * 16 + nlo;
    const float bv = bias ? bias[n] : 0.0f;
#pragma unroll
    for (int i = 0; i < 4; ++i) {
#pragma unroll
      for (int v = 0; v < 8; ++v) {
        const int m = m0 + wrow * 64 + i * 16 + v + 8 * half;
        float val = acc[i][j][v] + bv;
        if (relu) val = fmaxf(val, 0.0f);
        if (outF) outF[(size_t)m * N + n] = val;
        if (outB || outBT) {
          const u16 bb = f32_to_bf16_rne(val);
          if (outB) {
            if (!head_mode) {
              outB[(size_t)m * N + n] = bb;
            } else {  // [B,S,E] row m -> [B,H,S,HD]
              const int bb_ = m >> 11, s = m & (S_LEN - 1);
              const int h = n >> 6, d = n & 63;
              outB[(((size_t)(bb_ * H_NUM + h) * S_LEN + s) << 6) + d] = bb;
            }
          }
          if (outBT) {  // V^T: [B,H,HD,seq]
            const int bb_ = m >> 11, s = m & (S_LEN - 1);
            const int h = n >> 6, d = n & 63;
            outBT[((size_t)(bb_ * H_NUM + h) * HD_DIM + d) * S_LEN + s] = bb;
          }
        }
      }
    }
  }
}

// --------------------------- Flash attention -------------------------------
// grid = (S/64, B*H), block = 128 (4 waves). Wave owns 16 query rows.
// Q,K: [B,H,seq,64] bf16 (d-contiguous); Vt: [B,H,64,seq] bf16 (t-contiguous).
// Output: bf16 [B,S,E] (heads concatenated) feeding the O-projection GEMM.
__global__ __launch_bounds__(128)
void flash_attn_wmma(const u16* __restrict__ Q, const u16* __restrict__ Kb,
                     const u16* __restrict__ Vt, u16* __restrict__ Obf,
                     int S, int T, int causal)
{
  __shared__ __align__(16) u16 Ps[4 * 16 * 72];  // per-wave 16x64 P patch (+pad)

  const int tid  = threadIdx.x;
  const int lane = tid & 31;
  const int w    = tid >> 5;
  const int half = lane >> 4;
  const int mlo  = lane & 15;
  const int kh   = half * 16;       // A-fragment K half offset (contiguous)
  const int bh   = blockIdx.y;      // b*H + h
  const int b    = bh >> 4;
  const int h    = bh & 15;
  const int qblk = blockIdx.x;
  const int q0   = qblk * 64 + w * 16;
  const int qr   = q0 + mlo;        // this lane's A-matrix row

  union V16 { v16bf v; uint4 q[2]; };

  // Q fragments for both 32-wide d-steps (A-layout: 2x16B contiguous per lane)
  const size_t qbase = ((size_t)bh * S + qr) << 6;
  V16 aQ[2];
#pragma unroll
  for (int ds = 0; ds < 2; ++ds) {
    aQ[ds].q[0] = *(const uint4*)(Q + qbase + ds * 32 + kh);
    aQ[ds].q[1] = *(const uint4*)(Q + qbase + ds * 32 + 8 + kh);
  }

  float mrow[8], lrow[8];
  v8f o[4];
#pragma unroll
  for (int v = 0; v < 8; ++v) { mrow[v] = -3.0e38f; lrow[v] = 0.0f; }
#pragma unroll
  for (int d = 0; d < 4; ++d)
#pragma unroll
    for (int e = 0; e < 8; ++e) o[d][e] = 0.0f;

  const int nblk = causal ? (qblk + 1) : (T >> 6);
  for (int tb = 0; tb < nblk; ++tb) {
    // ---------------- scores S = Q * K^T (4 N-subtiles x 2 k-steps) --------
    v8f s[4];
#pragma unroll
    for (int ns = 0; ns < 4; ++ns)
#pragma unroll
      for (int e = 0; e < 8; ++e) s[ns][e] = 0.0f;
#pragma unroll
    for (int ds = 0; ds < 2; ++ds) {
#pragma unroll
      for (int ns = 0; ns < 4; ++ns) {
        const int kr = tb * 64 + ns * 16 + mlo;    // key row for this lane
        const v16bf bk = *(const v16bf*)(Kb + (((size_t)bh * T + kr) << 6) +
                                         ds * 32 + half * 16);
        s[ns] = __builtin_amdgcn_wmma_f32_16x16x32_bf16(
            false, aQ[ds].v, false, bk, (short)0, s[ns], false, false);
      }
    }
    // ---------------- scale + causal mask ----------------------------------
#pragma unroll
    for (int ns = 0; ns < 4; ++ns) {
      const int t = tb * 64 + ns * 16 + mlo;       // key col held by this lane
#pragma unroll
      for (int v = 0; v < 8; ++v) {
        float sv = s[ns][v] * 0.125f;              // 1/sqrt(64)
        if (causal && t > (q0 + v + 8 * half)) sv = -3.0e38f;
        s[ns][v] = sv;
      }
    }
    // ---------------- online softmax (rows replicated over 16 lanes) -------
#pragma unroll
    for (int v = 0; v < 8; ++v) {
      float mx = fmaxf(fmaxf(s[0][v], s[1][v]), fmaxf(s[2][v], s[3][v]));
#pragma unroll
      for (int msk = 1; msk <= 8; msk <<= 1)
        mx = fmaxf(mx, __shfl_xor(mx, msk, 32));
      const float newm = fmaxf(mrow[v], mx);
      const float corr = exp2f((mrow[v] - newm) * 1.44269504f);
      float ps = 0.0f;
#pragma unroll
      for (int ns = 0; ns < 4; ++ns) {
        const float p = exp2f((s[ns][v] - newm) * 1.44269504f);
        s[ns][v] = p;
        ps += p;
      }
#pragma unroll
      for (int msk = 1; msk <= 8; msk <<= 1)
        ps += __shfl_xor(ps, msk, 32);
      lrow[v] = lrow[v] * corr + ps;
      mrow[v] = newm;
#pragma unroll
      for (int d = 0; d < 4; ++d) o[d][v] *= corr;
    }
    // ---------------- P: C-layout -> A-layout via wave-private LDS ---------
#pragma unroll
    for (int ns = 0; ns < 4; ++ns)
#pragma unroll
      for (int v = 0; v < 8; ++v)
        Ps[w * 1152 + (v + 8 * half) * 72 + ns * 16 + mlo] =
            f32_to_bf16_rne(s[ns][v]);
    asm volatile("s_wait_dscnt 0x0" ::: "memory");  // wave-local RAW fence
    // ---------------- O += P * V (V^T rows are t-contiguous B-frags) -------
#pragma unroll
    for (int ks = 0; ks < 2; ++ks) {
      V16 aP;
      aP.q[0] = *(const uint4*)(&Ps[w * 1152 + mlo * 72 + ks * 32 + kh]);
      aP.q[1] = *(const uint4*)(&Ps[w * 1152 + mlo * 72 + ks * 32 + 8 + kh]);
#pragma unroll
      for (int d = 0; d < 4; ++d) {
        const int drow = d * 16 + mlo;
        const v16bf bv = *(const v16bf*)(Vt + ((size_t)bh * HD_DIM + drow) * T +
                                         tb * 64 + ks * 32 + half * 16);
        o[d] = __builtin_amdgcn_wmma_f32_16x16x32_bf16(
            false, aP.v, false, bv, (short)0, o[d], false, false);
      }
    }
  }
  // ---------------- normalize + store (concat heads, bf16) -----------------
#pragma unroll
  for (int d = 0; d < 4; ++d)
#pragma unroll
    for (int v = 0; v < 8; ++v) {
      const int qrow = q0 + v + 8 * half;
      const float val = o[d][v] / lrow[v];
      Obf[(((size_t)(b * S + qrow)) << 10) + h * 64 + d * 16 + mlo] =
          f32_to_bf16_rne(val);
    }
}

// --------------------- fused residual + LayerNorm --------------------------
__global__ __launch_bounds__(256)
void add_layernorm(const float* __restrict__ X, const float* __restrict__ R,
                   const float* __restrict__ g, const float* __restrict__ bt,
                   float* __restrict__ outF, u16* __restrict__ outB)
{
  __shared__ float red[2][8];
  const int row = blockIdx.x;
  const float* x = X + (size_t)row * E_DIM;
  const float* r = R + (size_t)row * E_DIM;
  float vals[4];
  float s = 0.0f, s2 = 0.0f;
#pragma unroll
  for (int i = 0; i < 4; ++i) {
    const int c = threadIdx.x + i * 256;
    const float v = x[c] + r[c];
    vals[i] = v; s += v; s2 += v * v;
  }
#pragma unroll
  for (int msk = 1; msk < 32; msk <<= 1) {
    s  += __shfl_xor(s,  msk, 32);
    s2 += __shfl_xor(s2, msk, 32);
  }
  const int wid = threadIdx.x >> 5, lane = threadIdx.x & 31;
  if (lane == 0) { red[0][wid] = s; red[1][wid] = s2; }
  __syncthreads();
  float ts = 0.0f, ts2 = 0.0f;
#pragma unroll
  for (int i = 0; i < 8; ++i) { ts += red[0][i]; ts2 += red[1][i]; }
  const float mean = ts * (1.0f / E_DIM);
  const float var  = ts2 * (1.0f / E_DIM) - mean * mean;
  const float inv  = rsqrtf(var + 1e-5f);
#pragma unroll
  for (int i = 0; i < 4; ++i) {
    const int c = threadIdx.x + i * 256;
    const float y = (vals[i] - mean) * inv * g[c] + bt[c];
    if (outF) outF[(size_t)row * E_DIM + c] = y;
    if (outB) outB[(size_t)row * E_DIM + c] = f32_to_bf16_rne(y);
  }
}

// ---------------------------------------------------------------------------
extern "C" void kernel_launch(void* const* d_in, const int* in_sizes, int n_in,
                              void* d_out, int out_size, void* d_ws, size_t ws_size,
                              hipStream_t stream) {
  (void)in_sizes; (void)n_in; (void)out_size; (void)ws_size;
  const float* x    = (const float*)d_in[0];
  const float* enc  = (const float*)d_in[1];
  const float* Wsa[4] = { (const float*)d_in[2], (const float*)d_in[4],
                          (const float*)d_in[6], (const float*)d_in[8] };
  const float* bsa[4] = { (const float*)d_in[3], (const float*)d_in[5],
                          (const float*)d_in[7], (const float*)d_in[9] };
  const float* Wca[4] = { (const float*)d_in[10], (const float*)d_in[12],
                          (const float*)d_in[14], (const float*)d_in[16] };
  const float* bca[4] = { (const float*)d_in[11], (const float*)d_in[13],
                          (const float*)d_in[15], (const float*)d_in[17] };
  const float* w1 = (const float*)d_in[18]; const float* b1 = (const float*)d_in[19];
  const float* w2 = (const float*)d_in[20]; const float* b2 = (const float*)d_in[21];
  const float* lng[3] = { (const float*)d_in[22], (const float*)d_in[24], (const float*)d_in[26] };
  const float* lnb[3] = { (const float*)d_in[23], (const float*)d_in[25], (const float*)d_in[27] };
  float* out = (float*)d_out;

  // ---- bump allocator over workspace (~208 MiB used) -----------------------
  char* ws = (char*)d_ws;
  size_t off = 0;
  auto alloc = [&](size_t bytes) -> void* {
    void* p = ws + off;
    off = (off + bytes + 255) & ~(size_t)255;
    return p;
  };
  const size_t EE  = (size_t)E_DIM * E_DIM;
  const size_t ME  = (size_t)M_ROWS * E_DIM;   // 4.19M elems
  const size_t MF  = (size_t)M_ROWS * F_DIM;

  u16 *wbsa[4], *wbca[4];
  for (int i = 0; i < 4; ++i) wbsa[i] = (u16*)alloc(EE * 2);
  for (int i = 0; i < 4; ++i) wbca[i] = (u16*)alloc(EE * 2);
  u16* w1b = (u16*)alloc((size_t)F_DIM * E_DIM * 2);
  u16* w2b = (u16*)alloc((size_t)E_DIM * F_DIM * 2);
  u16* xbf   = (u16*)alloc(ME * 2);
  u16* encbf = (u16*)alloc(ME * 2);
  u16 *q1 = (u16*)alloc(ME * 2), *k1 = (u16*)alloc(ME * 2),
      *vt1 = (u16*)alloc(ME * 2), *o1b = (u16*)alloc(ME * 2);
  u16 *q2 = (u16*)alloc(ME * 2), *k2 = (u16*)alloc(ME * 2),
      *vt2 = (u16*)alloc(ME * 2), *o2b = (u16*)alloc(ME * 2);
  float* tmpF = (float*)alloc(ME * 4);
  float* h1F  = (float*)alloc(ME * 4);
  float* h2F  = (float*)alloc(ME * 4);
  u16*   h1B  = (u16*)alloc(ME * 2);
  u16*   h2B  = (u16*)alloc(ME * 2);
  u16*   ffB  = (u16*)alloc(MF * 2);

  auto cvt = [&](const float* in, u16* o, size_t n) {
    cvt_f32_to_bf16<<<(unsigned)((n + 255) / 256), 256, 0, stream>>>(in, o, (int)n);
  };
  auto gemm = [&](const u16* A, const u16* W, const float* bias, float* oF,
                  u16* oB, u16* oBT, int M, int N, int K, int relu, int hm) {
    dim3 grid(N / 128, M / 128);
    gemm_bf16_wmma<<<grid, 256, 0, stream>>>(A, W, bias, oF, oB, oBT, M, N, K, relu, hm);
  };

  // ---- 1) bf16 conversions ------------------------------------------------
  for (int i = 0; i < 4; ++i) cvt(Wsa[i], wbsa[i], EE);
  for (int i = 0; i < 4; ++i) cvt(Wca[i], wbca[i], EE);
  cvt(w1, w1b, (size_t)F_DIM * E_DIM);
  cvt(w2, w2b, (size_t)E_DIM * F_DIM);
  cvt(x, xbf, ME);
  cvt(enc, encbf, ME);

  // ---- 2) self-attention --------------------------------------------------
  gemm(xbf, wbsa[0], bsa[0], nullptr, q1, nullptr, M_ROWS, E_DIM, E_DIM, 0, 1);
  gemm(xbf, wbsa[1], bsa[1], nullptr, k1, nullptr, M_ROWS, E_DIM, E_DIM, 0, 1);
  gemm(xbf, wbsa[2], bsa[2], nullptr, nullptr, vt1, M_ROWS, E_DIM, E_DIM, 0, 1);
  {
    dim3 ag(S_LEN / 64, B_NUM * H_NUM);
    flash_attn_wmma<<<ag, 128, 0, stream>>>(q1, k1, vt1, o1b, S_LEN, S_LEN, 1);
  }
  gemm(o1b, wbsa[3], bsa[3], tmpF, nullptr, nullptr, M_ROWS, E_DIM, E_DIM, 0, 0);
  add_layernorm<<<M_ROWS, 256, 0, stream>>>(tmpF, x, lng[0], lnb[0], h1F, h1B);

  // ---- 3) cross-attention -------------------------------------------------
  gemm(h1B,   wbca[0], bca[0], nullptr, q2, nullptr, M_ROWS, E_DIM, E_DIM, 0, 1);
  gemm(encbf, wbca[1], bca[1], nullptr, k2, nullptr, M_ROWS, E_DIM, E_DIM, 0, 1);
  gemm(encbf, wbca[2], bca[2], nullptr, nullptr, vt2, M_ROWS, E_DIM, E_DIM, 0, 1);
  {
    dim3 ag(S_LEN / 64, B_NUM * H_NUM);
    flash_attn_wmma<<<ag, 128, 0, stream>>>(q2, k2, vt2, o2b, S_LEN, T_LEN, 0);
  }
  gemm(o2b, wbca[3], bca[3], tmpF, nullptr, nullptr, M_ROWS, E_DIM, E_DIM, 0, 0);
  add_layernorm<<<M_ROWS, 256, 0, stream>>>(tmpF, h1F, lng[1], lnb[1], h2F, h2B);

  // ---- 4) FFN -------------------------------------------------------------
  gemm(h2B, w1b, b1, nullptr, ffB, nullptr, M_ROWS, F_DIM, E_DIM, 1, 0);
  gemm(ffB, w2b, b2, tmpF, nullptr, nullptr, M_ROWS, E_DIM, F_DIM, 0, 0);
  add_layernorm<<<M_ROWS, 256, 0, stream>>>(tmpF, h2F, lng[2], lnb[2], out, nullptr);
}
